// SwinTransformerBlock_37649683317325
// MI455X (gfx1250) — matrix operand order repaired and checked by reading, not compiled, not run on hardware
//
#include <hip/hip_runtime.h>

typedef __attribute__((ext_vector_type(16))) _Float16 v16h;
typedef __attribute__((ext_vector_type(4)))  _Float16 v4h;
typedef __attribute__((ext_vector_type(8)))  float    v8f;

#define T_TOK   200704      // 64*56*56 = 4096 windows * 49 tokens
#define C_DIM   192
#define NHEAD   6
#define HD      32
#define HIDDEN  768

// ---------------------------------------------------------------------------
// WMMA helpers. Fragments are kept in LDS in *register order*:
// each 16x16(xK32) operand block = 32 lanes x 16 halves, so a lane's fragment
// is 32 contiguous bytes -> 2x ds_load_b128.
// Layouts per CDNA5 ISA 05_wmma §7.12.2:
//  A 16x32: lane = (m&15) | ((k>>3)&1)<<4 ; half = (k&7) + ((k>>4)<<3)
//  B 32x16: lane = (n&15) | ((k>>3)&1)<<4 ; half = (k&7) + ((k>>4)<<3)
//  C/D f32: row = ((lane>>4)<<3)+v ; col = lane&15
// ---------------------------------------------------------------------------
__device__ __forceinline__ v8f wmma_f16(v16h a, v16h b, v8f c) {
  return __builtin_amdgcn_wmma_f32_16x16x32_f16(false, a, false, b, (short)0, c,
                                                false, false);
}
__device__ __forceinline__ int a_pos(int r, int k) {   // r,k within 16x32 block
  return ((r & 15) + (((k >> 3) & 1) << 4)) * 16 + (k & 7) + ((k >> 4) << 3);
}
__device__ __forceinline__ int b_pos(int k, int n) {   // k,n within 32x16 block
  return ((n & 15) + (((k >> 3) & 1) << 4)) * 16 + (k & 7) + ((k >> 4) << 3);
}
__device__ __forceinline__ v16h frag_at(const _Float16* blk) {
  return *(const v16h*)(blk + (threadIdx.x & 31) * 16);
}
__device__ __forceinline__ int d_row(int lane, int v) { return ((lane >> 4) << 3) + v; }
__device__ __forceinline__ int d_col(int lane)        { return lane & 15; }

// Two-phase staging: batch global loads into regs, then LDS stores, in groups
// of 4 iterations (128 threads). Counted loops -> no exec-mask while loops,
// 4 outstanding global loads before the first s_wait_loadcnt.
// NQ quads of 4 halves; quad q -> row = q>>3, k = (q&7)*4  (K-extent 32)
template <int NQ>
__device__ __forceinline__ void stage_a_f16(const _Float16* __restrict__ src,
                                            int ld, _Float16* dst, int tid) {
#pragma unroll
  for (int base = 0; base < NQ; base += 512) {
    v4h tmp[4];
#pragma unroll
    for (int it = 0; it < 4; ++it) {
      int q = base + it * 128 + tid;
      int r = q >> 3, k = (q & 7) * 4;
      tmp[it] = *(const v4h*)(src + (size_t)r * ld + k);
    }
#pragma unroll
    for (int it = 0; it < 4; ++it) {
      int q = base + it * 128 + tid;
      int r = q >> 3, k = (q & 7) * 4;
      *(v4h*)(dst + (r >> 4) * 512 + a_pos(r, k)) = tmp[it];
    }
  }
}
// NQ quads; quad q -> n = q>>3, k = (q&7)*4 ; converts f32 weights to f16
template <int NQ>
__device__ __forceinline__ void stage_b_f32(const float* __restrict__ src,
                                            int ld, _Float16* dst, int tid) {
#pragma unroll
  for (int base = 0; base < NQ; base += 512) {
    float4 tmp[4];
#pragma unroll
    for (int it = 0; it < 4; ++it) {
      int q = base + it * 128 + tid;
      int n = q >> 3, k = (q & 7) * 4;
      tmp[it] = *(const float4*)(src + (size_t)n * ld + k);
    }
#pragma unroll
    for (int it = 0; it < 4; ++it) {
      int q = base + it * 128 + tid;
      int n = q >> 3, k = (q & 7) * 4;
      v4h hv = {(_Float16)tmp[it].x, (_Float16)tmp[it].y,
                (_Float16)tmp[it].z, (_Float16)tmp[it].w};
      *(v4h*)(dst + (n >> 4) * 512 + b_pos(k, n)) = hv;
    }
  }
}

// ---------------------------------------------------------------------------
// Kernel 1: LN1 + cyclic shift(-3,-3) + window partition -> f16 [4096*49,192]
// ---------------------------------------------------------------------------
__global__ void __launch_bounds__(256)
ln1_shift_window_kernel(const float* __restrict__ x, const float* __restrict__ w,
                        const float* __restrict__ b, _Float16* __restrict__ xw) {
  int tok  = blockIdx.x * 8 + (threadIdx.x >> 5);
  int lane = threadIdx.x & 31;
  if (tok >= T_TOK) return;
  int win = tok / 49, t = tok - win * 49;
  int bb = win >> 6, wl = win & 63, wh = wl >> 3, ww = wl & 7;
  int i = t / 7, j = t - i * 7;
  int hs  = wh * 7 + i + 3; if (hs  >= 56) hs  -= 56;   // roll(-3): src = dst+3 mod 56
  int wsx = ww * 7 + j + 3; if (wsx >= 56) wsx -= 56;
  const float* src = x + (((size_t)bb * 56 + hs) * 56 + wsx) * C_DIM;
  float vals[6]; float s = 0.f, s2 = 0.f;
#pragma unroll
  for (int e = 0; e < 6; ++e) {
    vals[e] = src[lane + 32 * e];
    s += vals[e]; s2 += vals[e] * vals[e];
  }
#pragma unroll
  for (int off = 16; off > 0; off >>= 1) { s += __shfl_xor(s, off); s2 += __shfl_xor(s2, off); }
  float mean = s * (1.f / 192.f);
  float var  = s2 * (1.f / 192.f) - mean * mean;
  float rstd = rsqrtf(var + 1e-5f);
  _Float16* dst = xw + (size_t)tok * C_DIM;
#pragma unroll
  for (int e = 0; e < 6; ++e) {
    int c = lane + 32 * e;
    dst[c] = (_Float16)((vals[e] - mean) * rstd * w[c] + b[c]);
  }
}

// ---------------------------------------------------------------------------
// Kernel 2: QKV GEMM [200704,192]x[192,576]+bias -> scatter q(*scale)/k/v f16
// ---------------------------------------------------------------------------
__global__ void __launch_bounds__(128)
qkv_gemm_kernel(const _Float16* __restrict__ xw, const float* __restrict__ qkvw,
                const float* __restrict__ qkvb, _Float16* __restrict__ qb,
                _Float16* __restrict__ kb, _Float16* __restrict__ vb) {
  __shared__ alignas(32) _Float16 As[4 * 512];   // 4 m-blocks
  __shared__ alignas(32) _Float16 Bs[4 * 512];   // 4 n-blocks
  int mt = blockIdx.x / 9, nt = blockIdx.x % 9;
  int m0 = mt * 64, n0 = nt * 64;
  int tid = threadIdx.x, lane = tid & 31, wv = tid >> 5;
  v8f acc[4] = {};
  for (int k0 = 0; k0 < C_DIM; k0 += 32) {
    stage_a_f16<512>(xw + (size_t)m0 * C_DIM + k0, C_DIM, As, tid);
    stage_b_f32<512>(qkvw + (size_t)n0 * C_DIM + k0, C_DIM, Bs, tid);
    __syncthreads();
    v16h a = frag_at(As + wv * 512);
#pragma unroll
    for (int j = 0; j < 4; ++j) {
      v16h bf = frag_at(Bs + j * 512);
      acc[j] = wmma_f16(a, bf, acc[j]);
    }
    __syncthreads();
  }
  const float scale = 0.17677669529663687f;      // 32^-0.5
#pragma unroll
  for (int j = 0; j < 4; ++j)
#pragma unroll
    for (int v = 0; v < 8; ++v) {
      int m = m0 + wv * 16 + d_row(lane, v);
      int g = n0 + j * 16 + d_col(lane);
      float val = acc[j][v] + qkvb[g];
      int which = g / C_DIM, c = g % C_DIM;
      int head = c >> 5, d = c & 31;
      int win = m / 49, t = m - win * 49;
      size_t o = ((size_t)(win * NHEAD + head) * 49 + t) * HD + d;
      if (which == 0)      qb[o] = (_Float16)(val * scale);
      else if (which == 1) kb[o] = (_Float16)val;
      else                 vb[o] = (_Float16)val;
    }
}

// ---------------------------------------------------------------------------
// Kernel 3: per (window, head) attention, 49 padded to 64
// ---------------------------------------------------------------------------
__global__ void __launch_bounds__(128)
attn_kernel(const _Float16* __restrict__ qb, const _Float16* __restrict__ kbuf,
            const _Float16* __restrict__ vb, const float* __restrict__ rpb,
            _Float16* __restrict__ ao) {
  __shared__ alignas(32) _Float16 Qs[4 * 512];   // A blocks: 64 tok x 32 d
  __shared__ alignas(32) _Float16 Kb[4 * 512];   // B blocks: k=d(32), n=tok(64)
  __shared__ alignas(32) _Float16 Vb[4 * 512];   // B blocks: [kstep2][ntile2]
  __shared__ alignas(32) _Float16 Pa[8 * 512];   // A blocks: [mtile4][kstep2]
  __shared__ float Ss[64 * 68];
  __shared__ int   lab[49];
  int win = blockIdx.x / NHEAD, head = blockIdx.x % NHEAD;
  int tid = threadIdx.x, lane = tid & 31, wv = tid >> 5;
  size_t base = (size_t)(win * NHEAD + head) * 49 * HD;
  {  // two-phase gather of q/k/v tiles (4 quads each per thread)
    v4h qv[4], kv[4], vv[4];
#pragma unroll
    for (int it = 0; it < 4; ++it) {
      int q = it * 128 + tid;
      int t = q >> 3, d = (q & 7) * 4;
      v4h z = {};
      qv[it] = z; kv[it] = z; vv[it] = z;
      if (t < 49) {
        qv[it] = *(const v4h*)(qb   + base + t * HD + d);
        kv[it] = *(const v4h*)(kbuf + base + t * HD + d);
        vv[it] = *(const v4h*)(vb   + base + t * HD + d);
      }
    }
#pragma unroll
    for (int it = 0; it < 4; ++it) {
      int q = it * 128 + tid;
      int t = q >> 3, d = (q & 7) * 4;
      *(v4h*)(Qs + (t >> 4) * 512 + a_pos(t, d)) = qv[it];
      *(v4h*)(Kb + (t >> 4) * 512 + b_pos(d, t)) = kv[it];
      int kst = t >> 5, kk = t & 31;               // V: k=t, n=d -> scalar scatter
#pragma unroll
      for (int u = 0; u < 4; ++u)
        Vb[(kst * 2 + ((d + u) >> 4)) * 512 + b_pos(kk, d + u)] = vv[it][u];
    }
  }
  if (tid < 49) {  // shifted-window region labels (reproduces ATTN_MASK)
    int wl = win & 63, wh = wl >> 3, ww = wl & 7;
    int hc = wh * 7 + tid / 7, wc = ww * 7 + tid % 7;
    int rh = (hc < 49) ? 0 : (hc < 53 ? 1 : 2);
    int rw = (wc < 49) ? 0 : (wc < 53 ? 1 : 2);
    lab[tid] = rh * 3 + rw;
  }
  __syncthreads();
  {  // S = q.k^T + bias + mask
    v16h a = frag_at(Qs + wv * 512);
#pragma unroll
    for (int ntt = 0; ntt < 4; ++ntt) {
      v16h bf = frag_at(Kb + ntt * 512);
      v8f c = {};
      c = wmma_f16(a, bf, c);
#pragma unroll
      for (int v = 0; v < 8; ++v) {
        int r  = wv * 16 + d_row(lane, v);
        int cc = ntt * 16 + d_col(lane);
        float s = c[v];
        if (r < 49 && cc < 49) {
          int ri = r / 7, rj = r % 7, ci = cc / 7, cj = cc % 7;
          int rid = (ri - ci + 6) * 13 + (rj - cj + 6);  // REL_IDX formula
          s += rpb[rid * NHEAD + head];
          if (lab[r] != lab[cc]) s -= 100.f;
        } else {
          s = -1e30f;
        }
        Ss[r * 68 + cc] = s;
      }
    }
  }
  __syncthreads();
  if (tid < 64) {  // row softmax -> Pa (swizzled A layout, quad stores)
    int r = tid;
    if (r < 49) {
      float mx = -1e30f;
      for (int c = 0; c < 49; ++c) mx = fmaxf(mx, Ss[r * 68 + c]);
      float sum = 0.f;
      for (int c = 0; c < 49; ++c) sum += expf(Ss[r * 68 + c] - mx);
      float inv = 1.f / sum;
#pragma unroll
      for (int c = 0; c < 64; c += 4) {
        v4h pv;
#pragma unroll
        for (int u = 0; u < 4; ++u) {
          int cc = c + u;
          pv[u] = (cc < 49) ? (_Float16)(expf(Ss[r * 68 + cc] - mx) * inv)
                            : (_Float16)0.f;
        }
        *(v4h*)(Pa + ((r >> 4) * 2 + (c >> 5)) * 512 + a_pos(r, c & 31)) = pv;
      }
    } else {
#pragma unroll
      for (int c = 0; c < 64; c += 4) {
        v4h z = {};
        *(v4h*)(Pa + ((r >> 4) * 2 + (c >> 5)) * 512 + a_pos(r, c & 31)) = z;
      }
    }
  }
  __syncthreads();
  v8f o[2] = {};  // O = P.V (K=64 padded -> 2 steps)
#pragma unroll
  for (int ks = 0; ks < 2; ++ks) {
    v16h a = frag_at(Pa + (wv * 2 + ks) * 512);
#pragma unroll
    for (int ntt = 0; ntt < 2; ++ntt) {
      v16h bf = frag_at(Vb + (ks * 2 + ntt) * 512);
      o[ntt] = wmma_f16(a, bf, o[ntt]);
    }
  }
#pragma unroll
  for (int ntt = 0; ntt < 2; ++ntt)
#pragma unroll
    for (int v = 0; v < 8; ++v) {
      int r = wv * 16 + d_row(lane, v);
      int d = ntt * 16 + d_col(lane);
      if (r < 49)
        ao[((size_t)win * 49 + r) * C_DIM + head * HD + d] = (_Float16)o[ntt][v];
    }
}

// ---------------------------------------------------------------------------
// Kernel 4: proj GEMM + window-reverse + roll(+3,+3) + residual -> d_out (x2)
// ---------------------------------------------------------------------------
__global__ void __launch_bounds__(128)
proj_kernel(const _Float16* __restrict__ ao, const float* __restrict__ pw,
            const float* __restrict__ pb, const float* __restrict__ x,
            float* __restrict__ out) {
  __shared__ alignas(32) _Float16 As[4 * 512];
  __shared__ alignas(32) _Float16 Bs[4 * 512];
  int mt = blockIdx.x / 3, nt = blockIdx.x % 3;
  int m0 = mt * 64, n0 = nt * 64;
  int tid = threadIdx.x, lane = tid & 31, wv = tid >> 5;
  v8f acc[4] = {};
  for (int k0 = 0; k0 < C_DIM; k0 += 32) {
    stage_a_f16<512>(ao + (size_t)m0 * C_DIM + k0, C_DIM, As, tid);
    stage_b_f32<512>(pw + (size_t)n0 * C_DIM + k0, C_DIM, Bs, tid);
    __syncthreads();
    v16h a = frag_at(As + wv * 512);
#pragma unroll
    for (int j = 0; j < 4; ++j) {
      v16h bf = frag_at(Bs + j * 512);
      acc[j] = wmma_f16(a, bf, acc[j]);
    }
    __syncthreads();
  }
#pragma unroll
  for (int j = 0; j < 4; ++j)
#pragma unroll
    for (int v = 0; v < 8; ++v) {
      int m   = m0 + wv * 16 + d_row(lane, v);
      int cch = n0 + j * 16 + d_col(lane);
      int win = m / 49, t = m - win * 49;
      int bb = win >> 6, wl = win & 63, wh = wl >> 3, ww = wl & 7;
      int i = t / 7, jj = t - i * 7;
      int h0 = wh * 7 + i + 3;  if (h0 >= 56) h0 -= 56;   // roll back (+3)
      int w0 = ww * 7 + jj + 3; if (w0 >= 56) w0 -= 56;
      size_t o = (((size_t)bb * 56 + h0) * 56 + w0) * C_DIM + cch;
      out[o] = x[o] + acc[j][v] + pb[cch];
    }
}

// ---------------------------------------------------------------------------
// Kernel 5: LN2 over x2 (d_out) -> f16 token-major
// ---------------------------------------------------------------------------
__global__ void __launch_bounds__(256)
ln2_kernel(const float* __restrict__ x2, const float* __restrict__ w,
           const float* __restrict__ b, _Float16* __restrict__ mo) {
  int tok  = blockIdx.x * 8 + (threadIdx.x >> 5);
  int lane = threadIdx.x & 31;
  if (tok >= T_TOK) return;
  const float* src = x2 + (size_t)tok * C_DIM;
  float vals[6]; float s = 0.f, s2 = 0.f;
#pragma unroll
  for (int e = 0; e < 6; ++e) {
    vals[e] = src[lane + 32 * e];
    s += vals[e]; s2 += vals[e] * vals[e];
  }
#pragma unroll
  for (int off = 16; off > 0; off >>= 1) { s += __shfl_xor(s, off); s2 += __shfl_xor(s2, off); }
  float mean = s * (1.f / 192.f);
  float var  = s2 * (1.f / 192.f) - mean * mean;
  float rstd = rsqrtf(var + 1e-5f);
  _Float16* dst = mo + (size_t)tok * C_DIM;
#pragma unroll
  for (int e = 0; e < 6; ++e) {
    int c = lane + 32 * e;
    dst[c] = (_Float16)((vals[e] - mean) * rstd * w[c] + b[c]);
  }
}

// ---------------------------------------------------------------------------
// Kernel 6: fused MLP. 64-token tile; hidden looped in 64-chunks:
// fc1 WMMA -> exact GELU -> swizzled f16 chunk in LDS -> fc2 WMMA accumulate.
// ---------------------------------------------------------------------------
__global__ void __launch_bounds__(128)
mlp_fused_kernel(const _Float16* __restrict__ m16, const float* __restrict__ fc1w,
                 const float* __restrict__ fc1b, const float* __restrict__ fc2w,
                 const float* __restrict__ fc2b, float* __restrict__ out) {
  __shared__ alignas(32) _Float16 Ms[24 * 512];   // A: [mtile4][kstep6]
  __shared__ alignas(32) _Float16 W1s[4 * 512];   // B: 4 n-blocks
  __shared__ alignas(32) _Float16 Acts[8 * 512];  // A: [mtile4][kstep2]
  __shared__ alignas(32) _Float16 W2s[12 * 512];  // B: 12 n-blocks
  int tid = threadIdx.x, lane = tid & 31, wv = tid >> 5;
  size_t tb = (size_t)blockIdx.x * 64;
  // stage the 64x192 LN2 tile (6 groups of 4 quads/thread, two-phase)
#pragma unroll
  for (int grp = 0; grp < 6; ++grp) {
    v4h tmp[4];
#pragma unroll
    for (int it = 0; it < 4; ++it) {
      int q = grp * 512 + it * 128 + tid;
      int r = q / 48, c = (q - r * 48) * 4;
      tmp[it] = *(const v4h*)(m16 + (tb + r) * C_DIM + c);
    }
#pragma unroll
    for (int it = 0; it < 4; ++it) {
      int q = grp * 512 + it * 128 + tid;
      int r = q / 48, c = (q - r * 48) * 4;
      *(v4h*)(Ms + ((r >> 4) * 6 + (c >> 5)) * 512 + a_pos(r, c & 31)) = tmp[it];
    }
  }
  v8f co[12] = {};
  for (int hc = 0; hc < HIDDEN / 64; ++hc) {
    int h0 = hc * 64;
    v8f ca[4] = {};
    for (int ks = 0; ks < 6; ++ks) {
      int k0 = ks * 32;
      __syncthreads();
      stage_b_f32<512>(fc1w + (size_t)h0 * C_DIM + k0, C_DIM, W1s, tid);
      __syncthreads();
      v16h a = frag_at(Ms + (wv * 6 + ks) * 512);
#pragma unroll
      for (int j = 0; j < 4; ++j) {
        v16h bf = frag_at(W1s + j * 512);
        ca[j] = wmma_f16(a, bf, ca[j]);
      }
    }
#pragma unroll
    for (int j = 0; j < 4; ++j)          // bias + exact GELU -> Acts (own rows)
#pragma unroll
      for (int v = 0; v < 8; ++v) {
        int r = wv * 16 + d_row(lane, v);
        int c = j * 16 + d_col(lane);
        float z = ca[j][v] + fc1b[h0 + c];
        float g = 0.5f * z * (1.f + erff(z * 0.70710678118654752f));
        Acts[(wv * 2 + (c >> 5)) * 512 + a_pos(r, c & 31)] = (_Float16)g;
      }
    for (int ks = 0; ks < 2; ++ks) {     // fc2 accumulation over this chunk
      int k0 = ks * 32;
      __syncthreads();
      stage_b_f32<1536>(fc2w + h0 + k0, HIDDEN, W2s, tid);
      __syncthreads();
      v16h a = frag_at(Acts + (wv * 2 + ks) * 512);
#pragma unroll
      for (int j = 0; j < 12; ++j) {
        v16h bf = frag_at(W2s + j * 512);
        co[j] = wmma_f16(a, bf, co[j]);
      }
    }
  }
#pragma unroll
  for (int j = 0; j < 12; ++j)
#pragma unroll
    for (int v = 0; v < 8; ++v) {
      int r = wv * 16 + d_row(lane, v);
      int c = j * 16 + d_col(lane);
      size_t o = (tb + r) * C_DIM + c;
      out[o] = out[o] + co[j][v] + fc2b[c];  // residual x2 already in out
    }
}

// ---------------------------------------------------------------------------
extern "C" void kernel_launch(void* const* d_in, const int* in_sizes, int n_in,
                              void* d_out, int out_size, void* d_ws, size_t ws_size,
                              hipStream_t stream) {
  const float* x     = (const float*)d_in[0];
  const float* n1w   = (const float*)d_in[1];
  const float* n1b   = (const float*)d_in[2];
  const float* qkvw  = (const float*)d_in[3];
  const float* qkvb  = (const float*)d_in[4];
  const float* projw = (const float*)d_in[5];
  const float* projb = (const float*)d_in[6];
  const float* rpb   = (const float*)d_in[7];
  const float* n2w   = (const float*)d_in[8];
  const float* n2b   = (const float*)d_in[9];
  const float* fc1w  = (const float*)d_in[10];
  const float* fc1b  = (const float*)d_in[11];
  const float* fc2w  = (const float*)d_in[12];
  const float* fc2b  = (const float*)d_in[13];
  float* out = (float*)d_out;

  const size_t NE = (size_t)T_TOK * C_DIM;
  _Float16* buf0 = (_Float16*)d_ws;         // xw, later attn-out
  _Float16* buf1 = buf0 + NE;               // q, later LN2 output
  _Float16* buf2 = buf1 + NE;               // k
  _Float16* buf3 = buf2 + NE;               // v

  ln1_shift_window_kernel<<<T_TOK / 8, 256, 0, stream>>>(x, n1w, n1b, buf0);
  qkv_gemm_kernel<<<(T_TOK / 64) * 9, 128, 0, stream>>>(buf0, qkvw, qkvb, buf1, buf2, buf3);
  attn_kernel<<<4096 * NHEAD, 128, 0, stream>>>(buf1, buf2, buf3, rpb, buf0);
  proj_kernel<<<(T_TOK / 64) * 3, 128, 0, stream>>>(buf0, projw, projb, x, out);
  ln2_kernel<<<T_TOK / 8, 256, 0, stream>>>(out, n2w, n2b, buf1);
  mlp_fused_kernel<<<T_TOK / 64, 128, 0, stream>>>(buf1, fc1w, fc1b, fc2w, fc2b, out);
}
